// CASST_26113401159803
// MI455X (gfx1250) — compile-verified, matrix operand order checked by reading, not run on hardware
//
#include <hip/hip_runtime.h>
#include <hip/hip_bf16.h>
#include <math.h>

// ---------------------------------------------------------------------------
// CASST forward for MI455X (gfx1250, wave32). All large GEMMs run on
// v_wmma_f32_16x16x32_f16. A is LDS-staged with branch-free b128 copies
// (double buffered, one barrier per K-step). B operands are pre-swizzled in
// memory into WMMA fragment order so each wave loads its B fragments directly
// from global as coalesced b128s -- no LDS, no bounds checks in the hot loop.
// ---------------------------------------------------------------------------

typedef _Float16 h16;
typedef __attribute__((ext_vector_type(16))) _Float16 v16h;
typedef __attribute__((ext_vector_type(8)))  _Float16 v8h;
typedef __attribute__((ext_vector_type(8)))  float    v8f;

constexpr int kB     = 128;
constexpr int kBands = 30;
constexpr int kHW    = 15;
constexpr int kPix   = 225;
constexpr int kDim   = 512;
constexpr int kNH    = 8;
constexpr int kHD    = 64;
constexpr int kNTX   = 226;             // spatial tokens (+cls)
constexpr int kNTY   = 31;              // spectral tokens (+cls)
constexpr int kMX    = kB * kNTX;       // 28928 (divisible by 128)
constexpr int kMY    = kB * kNTY;       // 3968
constexpr int kKconv = 288;             // 270 zero-padded to 32-multiple
constexpr int kNTPX  = 256;             // padded token count, x branch
constexpr int kNTPY  = 64;              // padded token count, y branch

constexpr int TBM  = 128;
constexpr int TBN  = 64;
constexpr int TBK  = 32;
constexpr int TBKP = 40;                // padded LDS row stride (halves)
constexpr size_t SLACK = 1u << 17;      // per-allocation overread slack

enum { EPI_NONE = 0, EPI_BIAS = 1, EPI_BIAS_RELU = 2, EPI_BIAS_GELU = 3, EPI_SS_RELU = 4 };

// ---------------------------------------------------------------------------
// WMMA GEMM. A: f16 (M x Kp row-major, lda mult of 8, Kp mult of 32; rows
// beyond M readable garbage). Bsw: fragment-swizzled B, blocks of 512 halves
// indexed [kt*NB16 + n16]. C guarded stores. 256 thr = 8 waves; each wave
// computes a 16x64 strip of the 128x64 tile (4 accumulators).
// ---------------------------------------------------------------------------
__global__ __launch_bounds__(256)
void k_gemm(const h16* __restrict__ A, long long lda, long long sA,
            const h16* __restrict__ Bsw, long long sB, int NB16,
            float* __restrict__ Cf, long long ldc, long long sC,
            h16* __restrict__ Ch,
            int M, int N, int Kp, float alpha, int epi,
            const float* __restrict__ p0, const float* __restrict__ p1)
{
    __shared__ h16 lA[2 * TBM * TBKP];

    const int tid    = threadIdx.x;
    const int lane   = tid & 31;
    const int wave   = tid >> 5;        // 16-row strip
    const int half16 = lane >> 4;
    const int l15    = lane & 15;

    const int bt = blockIdx.z;
    const int m0 = blockIdx.y * TBM;
    const int n0 = blockIdx.x * TBN;
    const h16* Ab = A + (size_t)bt * sA;
    const h16* Bb = Bsw + (size_t)bt * sB + ((size_t)(n0 >> 4)) * 512 + lane * 16;

    const int smr = tid >> 1;           // staging row
    const int skc = (tid & 1) << 4;     // staging col (halves)

    v8f acc[4];
    acc[0] = {}; acc[1] = {}; acc[2] = {}; acc[3] = {};

    const int nkt = Kp >> 5;
    for (int kt = 0; kt < nkt; ++kt) {
        // ---- stage A tile (128x32), branch-free b128 copies ----
        h16* ls = lA + (kt & 1) * (TBM * TBKP) + smr * TBKP + skc;
        const h16* gs = Ab + (size_t)(m0 + smr) * lda + (kt << 5) + skc;
        *(v8h*)ls       = *(const v8h*)gs;
        *(v8h*)(ls + 8) = *(const v8h*)(gs + 8);
        __syncthreads();

        // ---- A fragment (two contiguous b128 chunks) ----
        const h16* ap = lA + (kt & 1) * (TBM * TBKP)
                      + (wave * 16 + l15) * TBKP + (half16 << 3);
        v8h a0 = *(const v8h*)ap;
        v8h a1 = *(const v8h*)(ap + 16);
        v16h af = __builtin_shufflevector(a0, a1, 0, 1, 2, 3, 4, 5, 6, 7,
                                          8, 9, 10, 11, 12, 13, 14, 15);

        // ---- B fragments direct from global (swizzled), 4 WMMAs ----
        const h16* bp = Bb + (size_t)kt * NB16 * 512;
        #pragma unroll
        for (int nt = 0; nt < 4; ++nt) {
            v16h bf = *(const v16h*)(bp + nt * 512);
            acc[nt] = __builtin_amdgcn_wmma_f32_16x16x32_f16(false, af, false, bf,
                                                             (short)0, acc[nt],
                                                             false, false);
        }
    }

    // ---- epilogue + guarded stores ----
    float* Cb = Cf ? Cf + (size_t)bt * sC : nullptr;
    h16*  Chb = Ch ? Ch + (size_t)bt * sC : nullptr;
    #pragma unroll
    for (int nt = 0; nt < 4; ++nt) {
        int gn = n0 + nt * 16 + l15;
        #pragma unroll
        for (int r = 0; r < 8; ++r) {
            int gm = m0 + wave * 16 + r + half16 * 8;
            if (gm < M && gn < N) {
                float v = acc[nt][r] * alpha;
                switch (epi) {
                    case EPI_BIAS:      v += p0[gn]; break;
                    case EPI_BIAS_RELU: v += p0[gn]; v = fmaxf(v, 0.f); break;
                    case EPI_BIAS_GELU: v += p0[gn];
                                        v = 0.5f * v * (1.f + erff(v * 0.70710678118654752f));
                                        break;
                    case EPI_SS_RELU:   v = v * p0[gn] + p1[gn]; v = fmaxf(v, 0.f); break;
                    default: break;
                }
                size_t o = (size_t)gm * ldc + gn;
                if (Cb)  Cb[o]  = v;
                if (Chb) Chb[o] = (h16)v;
            }
        }
    }
}

// ---------------------------------------------------------------------------
// Elementwise / reduction kernels
// ---------------------------------------------------------------------------

// W (N x Kreal row-major f32) -> fragment-swizzled f16 B (Kpad x Npad), pads=0
__global__ void k_cvt_sw(const float* __restrict__ W, h16* __restrict__ out,
                         int N, int Npad, int Kreal, int Kpad)
{
    long long i = (long long)blockIdx.x * 256 + threadIdx.x;
    if (i >= (long long)Kpad * Npad) return;
    int k = (int)(i / Npad), n = (int)(i % Npad);
    float val = (n < N && k < Kreal) ? W[(size_t)n * Kreal + k] : 0.f;
    int kt = k >> 5, kr = k & 31, nt = n >> 4, nr = n & 15;
    int half = (kr >> 3) & 1;
    int vv = ((kr >> 4) << 2) | ((kr >> 1) & 3);
    int parity = kr & 1;
    int ln = (half << 4) | nr;
    out[((size_t)kt * (Npad >> 4) + nt) * 512 + ln * 16 + vv * 2 + parity] = (h16)val;
}

__global__ void k_zero16(h16* __restrict__ p, long long n)
{
    long long i = (long long)blockIdx.x * 256 + threadIdx.x;
    if (i < n) p[i] = (h16)0.f;
}

// fold conv-bias + BN into per-channel scale/shift
__global__ void k_bnprep(const float* g, const float* be, const float* mn,
                         const float* vr, const float* cb,
                         float* scale, float* shift, int C)
{
    int i = blockIdx.x * 256 + threadIdx.x;
    if (i >= C) return;
    float inv = g[i] * rsqrtf(vr[i] + 1e-5f);
    scale[i] = inv;
    shift[i] = (cb[i] - mn[i]) * inv + be[i];
}

// per-band conv(1->128,3x3)+BN+ReLU+avgpool, f16 out (3840 x 128)
__global__ __launch_bounds__(128)
void k_cnnband(const float* __restrict__ x, const float* __restrict__ w,
               const float* __restrict__ scale, const float* __restrict__ shift,
               h16* __restrict__ out)
{
    __shared__ float tile[kPix];
    int bb = blockIdx.x, tid = threadIdx.x;
    for (int j = tid; j < kPix; j += 128) tile[j] = x[(size_t)bb * kPix + j];
    __syncthreads();
    int och = tid;
    float wr[9];
    #pragma unroll
    for (int t = 0; t < 9; ++t) wr[t] = w[och * 9 + t];
    float sc = scale[och], sh = shift[och];
    float pooled = 0.f;
    for (int p = 0; p < kPix; ++p) {
        int oy = p / kHW, ox = p % kHW;
        float s = 0.f;
        #pragma unroll
        for (int t = 0; t < 9; ++t) {
            int iy = oy + t / 3 - 1, ix = ox + t % 3 - 1;
            if (iy >= 0 && iy < kHW && ix >= 0 && ix < kHW) s += wr[t] * tile[iy * kHW + ix];
        }
        pooled += fmaxf(s * sc + sh, 0.f);
    }
    out[(size_t)bb * 128 + och] = (h16)(pooled * (1.f / 225.f));
}

// im2col for conv_h: A (28800 x 288) f16, cols 270..287 zero
__global__ void k_im2col(const float* __restrict__ x, h16* __restrict__ A)
{
    long long i = (long long)blockIdx.x * 256 + threadIdx.x;
    long long tot = (long long)kB * kPix * kKconv;
    if (i >= tot) return;
    long long r = i / kKconv; int k = (int)(i % kKconv);
    float v = 0.f;
    if (k < 270) {
        int b = (int)(r / kPix), p = (int)(r % kPix);
        int oy = p / kHW, ox = p % kHW;
        int c = k / 9, t = k % 9;
        int iy = oy + t / 3 - 1, ix = ox + t % 3 - 1;
        if (iy >= 0 && iy < kHW && ix >= 0 && ix < kHW)
            v = x[((size_t)b * kBands + c) * kPix + iy * kHW + ix];
    }
    A[i] = (h16)v;
}

__global__ void k_asm_spa(const float* __restrict__ tmp, const float* __restrict__ cls,
                          const float* __restrict__ pos, float* __restrict__ xs)
{
    long long i = (long long)blockIdx.x * 256 + threadIdx.x;
    if (i >= (long long)kMX * kDim) return;
    long long r = i / kDim; int d = (int)(i % kDim);
    int b = (int)(r / kNTX), t = (int)(r % kNTX);
    float v = (t == 0) ? cls[d] : tmp[((size_t)b * kPix + (t - 1)) * kDim + d];
    xs[i] = v + pos[(size_t)t * kDim + d];
}

__global__ void k_asm_spe(const float* __restrict__ f2, const float* __restrict__ cls,
                          const float* __restrict__ pos, float* __restrict__ ys)
{
    long long i = (long long)blockIdx.x * 256 + threadIdx.x;
    if (i >= (long long)kMY * kDim) return;
    long long r = i / kDim; int d = (int)(i % kDim);
    int b = (int)(r / kNTY), t = (int)(r % kNTY);
    float v = (t == 0) ? cls[d] : f2[((size_t)b * kBands + (t - 1)) * kDim + d];
    ys[i] = v + pos[(size_t)t * kDim + d];
}

// LayerNorm over 512 (one wave per row). clsSrc!=null: row t==0 of each sample
// reads the other branch's cls row.
__global__ __launch_bounds__(256)
void k_ln(const float* __restrict__ src, const float* __restrict__ clsSrc,
          long long clsStride, int Ntok, int nrows,
          const float* __restrict__ w, const float* __restrict__ b,
          h16* __restrict__ out)
{
    int row  = blockIdx.x * 8 + (threadIdx.x >> 5);
    int lane = threadIdx.x & 31;
    if (row >= nrows) return;
    int bidx = row / Ntok, t = row % Ntok;
    const float* p = (t == 0 && clsSrc) ? (clsSrc + (size_t)bidx * clsStride)
                                        : (src + (size_t)row * kDim);
    float s = 0.f, q = 0.f, vals[16];
    #pragma unroll
    for (int j = 0; j < 16; ++j) {
        float x = p[lane + j * 32];
        vals[j] = x; s += x; q += x * x;
    }
    #pragma unroll
    for (int o = 16; o >= 1; o >>= 1) { s += __shfl_xor(s, o); q += __shfl_xor(q, o); }
    float mean = s * (1.f / 512.f);
    float rstd = rsqrtf(q * (1.f / 512.f) - mean * mean + 1e-5f);
    h16* po = out + (size_t)row * kDim;
    #pragma unroll
    for (int j = 0; j < 16; ++j) {
        int d = lane + j * 32;
        po[d] = (h16)((vals[j] - mean) * rstd * w[d] + b[d]);
    }
}

// qkv f32 (M x 1536) -> q plain (bh, NtP, 64); k swizzled as trans-B
// (n=t,k=d); v swizzled as B (k=t,n=d). v buffer must be pre-zeroed.
__global__ void k_split_qkv(const float* __restrict__ qkv, h16* __restrict__ q,
                            h16* __restrict__ kk, h16* __restrict__ vv16,
                            int Ntok, int NtP, long long M)
{
    long long i = (long long)blockIdx.x * 256 + threadIdx.x;
    if (i >= M * 1536) return;
    long long r = i / 1536; int c = (int)(i % 1536);
    int which = c >> 9; int hc = c & 511; int h = hc >> 6; int d = hc & 63;
    int b = (int)(r / Ntok), t = (int)(r % Ntok);
    h16 val = (h16)qkv[i];
    size_t bo = (size_t)(b * kNH + h) * NtP * kHD;
    if (which == 0) {
        q[bo + (size_t)t * kHD + d] = val;
    } else if (which == 1) {              // K^T fragment layout: kdim=d, ndim=t
        int kt = d >> 5, kr = d & 31, nt = t >> 4, nr = t & 15;
        int half = (kr >> 3) & 1, vv = ((kr >> 4) << 2) | ((kr >> 1) & 3), pa = kr & 1;
        kk[bo + ((size_t)kt * (NtP >> 4) + nt) * 512 + ((half << 4) | nr) * 16 + vv * 2 + pa] = val;
    } else {                              // V fragment layout: kdim=t, ndim=d
        int kt = t >> 5, kr = t & 31, nt = d >> 4, nr = d & 15;
        int half = (kr >> 3) & 1, vv = ((kr >> 4) << 2) | ((kr >> 1) & 3), pa = kr & 1;
        vv16[bo + ((size_t)kt * 4 + nt) * 512 + ((half << 4) | nr) * 16 + vv * 2 + pa] = val;
    }
}

// rowwise softmax; input stride NtP, writes f16 probs (stride NtP) with
// zero-padded columns [Nt, NtP)
__global__ __launch_bounds__(256)
void k_softmax(const float* __restrict__ sc, h16* __restrict__ probs,
               int Nt, int NtP, long long rows)
{
    long long row = (long long)blockIdx.x * 8 + (threadIdx.x >> 5);
    int lane = threadIdx.x & 31;
    if (row >= rows) return;
    long long bt = row / Nt, t = row % Nt;
    const float* p = sc + (bt * NtP + t) * NtP;
    float mx = -3.4e38f;
    for (int j = lane; j < Nt; j += 32) mx = fmaxf(mx, p[j]);
    #pragma unroll
    for (int o = 16; o >= 1; o >>= 1) mx = fmaxf(mx, __shfl_xor(mx, o));
    float sum = 0.f;
    for (int j = lane; j < Nt; j += 32) sum += __expf(p[j] - mx);
    #pragma unroll
    for (int o = 16; o >= 1; o >>= 1) sum += __shfl_xor(sum, o);
    float inv = 1.f / sum;
    h16* po = probs + (bt * NtP + t) * NtP;
    for (int j = lane; j < NtP; j += 32)
        po[j] = (j < Nt) ? (h16)(__expf(p[j] - mx) * inv) : (h16)0.f;
}

// attout (bh, NtP, 64) f32 -> merged (M x 512) f16
__global__ void k_merge(const float* __restrict__ att, h16* __restrict__ out,
                        int Ntok, int NtP, long long M)
{
    long long i = (long long)blockIdx.x * 256 + threadIdx.x;
    if (i >= M * kDim) return;
    long long r = i / kDim; int c = (int)(i % kDim);
    int h = c >> 6, d = c & 63;
    int b = (int)(r / Ntok), t = (int)(r % Ntok);
    out[i] = (h16)att[((size_t)(b * kNH + h) * NtP + t) * kHD + d];
}

// cross-cls residual: row0: dst = 2*dst + other_proj[cls]; else dst += self_proj
__global__ void k_cross(float* __restrict__ dst, const float* __restrict__ selfp,
                        const float* __restrict__ otherp, int Nt, int NtO,
                        long long tot)
{
    long long i = (long long)blockIdx.x * 256 + threadIdx.x;
    if (i >= tot) return;
    long long r = i / kDim; int d = (int)(i % kDim);
    int b = (int)(r / Nt), t = (int)(r % Nt);
    float x = dst[i];
    if (t == 0) dst[i] = 2.f * x + otherp[(size_t)b * NtO * kDim + d];
    else        dst[i] = x + selfp[i];
}

__global__ void k_add(float* __restrict__ dst, const float* __restrict__ src, long long n)
{
    long long i = (long long)blockIdx.x * 256 + threadIdx.x;
    if (i < n) dst[i] += src[i];
}

// final cls LN + concat -> f16 (128 x 1024)
__global__ __launch_bounds__(256)
void k_headprep(const float* __restrict__ xs, const float* __restrict__ ys,
                const float* __restrict__ w1, const float* __restrict__ b1,
                const float* __restrict__ w2, const float* __restrict__ b2,
                h16* __restrict__ out)
{
    int item = blockIdx.x * 8 + (threadIdx.x >> 5);
    int lane = threadIdx.x & 31;
    if (item >= kB * 2) return;
    int b = item >> 1, which = item & 1;
    const float* p = which ? (ys + (size_t)b * kNTY * kDim) : (xs + (size_t)b * kNTX * kDim);
    const float* w = which ? w2 : w1;
    const float* bi = which ? b2 : b1;
    float s = 0.f, q = 0.f, vals[16];
    #pragma unroll
    for (int j = 0; j < 16; ++j) {
        float x = p[lane + j * 32];
        vals[j] = x; s += x; q += x * x;
    }
    #pragma unroll
    for (int o = 16; o >= 1; o >>= 1) { s += __shfl_xor(s, o); q += __shfl_xor(q, o); }
    float mean = s * (1.f / 512.f);
    float rstd = rsqrtf(q * (1.f / 512.f) - mean * mean + 1e-5f);
    h16* po = out + (size_t)b * 1024 + which * kDim;
    #pragma unroll
    for (int j = 0; j < 16; ++j) {
        int d = lane + j * 32;
        po[d] = (h16)((vals[j] - mean) * rstd * w[d] + bi[d]);
    }
}

// ---------------------------------------------------------------------------
static inline unsigned gdiv(long long n) { return (unsigned)((n + 255) / 256); }

static void gemm(hipStream_t s,
                 const h16* A, long long lda, long long sA,
                 const h16* Bsw, long long sB, int NB16,
                 float* Cf, h16* Ch, long long ldc, long long sC,
                 int M, int N, int Kp, int batch, float alpha, int epi,
                 const float* p0, const float* p1)
{
    dim3 grid((N + TBN - 1) / TBN, (M + TBM - 1) / TBM, batch);
    k_gemm<<<grid, 256, 0, s>>>(A, lda, sA, Bsw, sB, NB16,
                                Cf, ldc, sC, Ch, M, N, Kp, alpha, epi, p0, p1);
}

extern "C" void kernel_launch(void* const* d_in, const int* in_sizes, int n_in,
                              void* d_out, int out_size, void* d_ws, size_t ws_size,
                              hipStream_t stream)
{
    (void)in_sizes; (void)n_in; (void)out_size; (void)ws_size;

    const float* x         = (const float*)d_in[0];
    const float* conv_h_w  = (const float*)d_in[1];
    const float* conv_h_b  = (const float*)d_in[2];
    const float* bn_h_g    = (const float*)d_in[3];
    const float* bn_h_b    = (const float*)d_in[4];
    const float* bn_h_m    = (const float*)d_in[5];
    const float* bn_h_v    = (const float*)d_in[6];
    const float* cnn_conv_w= (const float*)d_in[7];
    const float* cnn_conv_b= (const float*)d_in[8];
    const float* cnn_bn_g  = (const float*)d_in[9];
    const float* cnn_bn_b  = (const float*)d_in[10];
    const float* cnn_bn_m  = (const float*)d_in[11];
    const float* cnn_bn_v  = (const float*)d_in[12];
    const float* cnn_fc_w  = (const float*)d_in[13];
    const float* cnn_fc_b  = (const float*)d_in[14];
    const float* spa_cls   = (const float*)d_in[15];
    const float* spe_cls   = (const float*)d_in[16];
    const float* spa_pos   = (const float*)d_in[17];
    const float* spe_pos   = (const float*)d_in[18];
    const float* n1_w      = (const float*)d_in[19];
    const float* n1_b      = (const float*)d_in[20];
    const float* n2_w      = (const float*)d_in[21];
    const float* n2_b      = (const float*)d_in[22];
    const float* qkv_w     = (const float*)d_in[23];
    const float* qkv_b     = (const float*)d_in[24];
    const float* proj_w    = (const float*)d_in[25];
    const float* proj_b    = (const float*)d_in[26];
    const float* fc1_w     = (const float*)d_in[27];
    const float* fc1_b     = (const float*)d_in[28];
    const float* fc2_w     = (const float*)d_in[29];
    const float* fc2_b     = (const float*)d_in[30];
    const float* norm1_w   = (const float*)d_in[31];
    const float* norm1_b   = (const float*)d_in[32];
    const float* norm2_w   = (const float*)d_in[33];
    const float* norm2_b   = (const float*)d_in[34];
    const float* head_w    = (const float*)d_in[35];
    const float* head_b    = (const float*)d_in[36];
    float* out = (float*)d_out;

    // ---- workspace layout ----
    char* wsp = (char*)d_ws;
    size_t off = 0;
    auto ALLOC = [&](size_t bytes) -> void* {
        void* p = wsp + off;
        off += (bytes + SLACK + 255) & ~(size_t)255;
        return p;
    };
    h16* wB_convh = (h16*)ALLOC((size_t)kKconv * 512 * sizeof(h16));
    h16* wB_cnnfc = (h16*)ALLOC((size_t)128 * 512 * sizeof(h16));
    h16* wB_qkv   = (h16*)ALLOC((size_t)2 * 512 * 1536 * sizeof(h16));
    h16* wB_proj  = (h16*)ALLOC((size_t)2 * 512 * 512 * sizeof(h16));
    h16* wB_fc1   = (h16*)ALLOC((size_t)2 * 512 * 512 * sizeof(h16));
    h16* wB_fc2   = (h16*)ALLOC((size_t)2 * 512 * 512 * sizeof(h16));
    h16* wB_head  = (h16*)ALLOC((size_t)1024 * 64 * sizeof(h16));
    float* sc_convh = (float*)ALLOC(512 * 4);
    float* sh_convh = (float*)ALLOC(512 * 4);
    float* sc_cnn   = (float*)ALLOC(128 * 4);
    float* sh_cnn   = (float*)ALLOC(128 * 4);
    h16*  f16band = (h16*)ALLOC((size_t)3840 * 128 * sizeof(h16));
    float* f2buf  = (float*)ALLOC((size_t)3840 * 512 * 4);
    float* xs     = (float*)ALLOC((size_t)kMX * kDim * 4);
    float* ys     = (float*)ALLOC((size_t)kMY * kDim * 4);
    h16*  A16a    = (h16*)ALLOC((size_t)kMX * kDim * sizeof(h16));
    h16*  A16b    = (h16*)ALLOC((size_t)kMX * kDim * sizeof(h16));
    float* bigf   = (float*)ALLOC((size_t)1024 * kNTPX * kNTPX * 4); // qkv/scores/attout
    h16*  q16     = (h16*)ALLOC((size_t)1024 * kNTPX * kHD * sizeof(h16));
    h16*  k16     = (h16*)ALLOC((size_t)1024 * kNTPX * kHD * sizeof(h16));
    h16*  v16     = (h16*)ALLOC((size_t)1024 * kNTPX * kHD * sizeof(h16));
    h16*  probs16 = (h16*)ALLOC((size_t)1024 * kNTPX * kNTPX * sizeof(h16));
    float* xproj  = (float*)ALLOC((size_t)kMX * kDim * 4);           // conv tmp / mlp scratch
    float* yproj  = (float*)ALLOC((size_t)kMY * kDim * 4);
    h16*  Ahead   = (h16*)ALLOC((size_t)kB * 1024 * sizeof(h16));

    // ---- weights -> fragment-swizzled f16 ----
    k_cvt_sw<<<gdiv((long long)kKconv*512), 256, 0, stream>>>(conv_h_w, wB_convh, 512, 512, 270, kKconv);
    k_cvt_sw<<<gdiv(128LL*512), 256, 0, stream>>>(cnn_fc_w, wB_cnnfc, 512, 512, 128, 128);
    for (int i = 0; i < 2; ++i) {
        k_cvt_sw<<<gdiv(512LL*1536), 256, 0, stream>>>(qkv_w + (size_t)i*1536*512,
                                                       wB_qkv + (size_t)i*512*1536, 1536, 1536, 512, 512);
        k_cvt_sw<<<gdiv(512LL*512), 256, 0, stream>>>(proj_w + (size_t)i*512*512,
                                                      wB_proj + (size_t)i*512*512, 512, 512, 512, 512);
        k_cvt_sw<<<gdiv(512LL*512), 256, 0, stream>>>(fc1_w + (size_t)i*512*512,
                                                      wB_fc1 + (size_t)i*512*512, 512, 512, 512, 512);
        k_cvt_sw<<<gdiv(512LL*512), 256, 0, stream>>>(fc2_w + (size_t)i*512*512,
                                                      wB_fc2 + (size_t)i*512*512, 512, 512, 512, 512);
    }
    k_cvt_sw<<<gdiv(1024LL*64), 256, 0, stream>>>(head_w, wB_head, 16, 64, 1024, 1024);
    k_bnprep<<<2, 256, 0, stream>>>(bn_h_g, bn_h_b, bn_h_m, bn_h_v, conv_h_b,
                                    sc_convh, sh_convh, 512);
    k_bnprep<<<1, 256, 0, stream>>>(cnn_bn_g, cnn_bn_b, cnn_bn_m, cnn_bn_v, cnn_conv_b,
                                    sc_cnn, sh_cnn, 128);

    // ---- spectral branch ----
    k_cnnband<<<kB * kBands, 128, 0, stream>>>(x, cnn_conv_w, sc_cnn, sh_cnn, f16band);
    gemm(stream, f16band, 128, 0, wB_cnnfc, 0, 32, f2buf, nullptr, 512, 0,
         kB * kBands, 512, 128, 1, 1.f, EPI_BIAS_RELU, cnn_fc_b, nullptr);
    k_asm_spe<<<gdiv((long long)kMY * kDim), 256, 0, stream>>>(f2buf, spe_cls, spe_pos, ys);

    // ---- spatial branch ----
    k_im2col<<<gdiv((long long)kB * kPix * kKconv), 256, 0, stream>>>(x, A16a);
    gemm(stream, A16a, kKconv, 0, wB_convh, 0, 32, xproj, nullptr, 512, 0,
         kB * kPix, 512, kKconv, 1, 1.f, EPI_SS_RELU, sc_convh, sh_convh);
    k_asm_spa<<<gdiv((long long)kMX * kDim), 256, 0, stream>>>(xproj, spa_cls, spa_pos, xs);

    // ---- transformer blocks ----
    auto attn_branch = [&](const float* src, const float* clsSrc, long long clsStride,
                           int Nt, int NtP, int Mrows, int i, float* projOut) {
        const int batch = kB * kNH;
        const int nb16 = NtP >> 4;
        k_ln<<<(Mrows + 7) / 8, 256, 0, stream>>>(src, clsSrc, clsStride, Nt, Mrows,
                                                  n1_w + i * 512, n1_b + i * 512, A16a);
        gemm(stream, A16a, 512, 0, wB_qkv + (size_t)i * 512 * 1536, 0, 96,
             bigf, nullptr, 1536, 0, Mrows, 1536, 512, 1, 1.f, EPI_BIAS,
             qkv_b + i * 1536, nullptr);
        k_zero16<<<gdiv((long long)batch * NtP * kHD), 256, 0, stream>>>(v16, (long long)batch * NtP * kHD);
        k_split_qkv<<<gdiv((long long)Mrows * 1536), 256, 0, stream>>>(bigf, q16, k16, v16,
                                                                       Nt, NtP, Mrows);
        // scores = 0.125 * Q K^T  (K pre-swizzled as trans-B)
        gemm(stream, q16, 64, (long long)NtP * kHD, k16, (long long)NtP * kHD, nb16,
             bigf, nullptr, NtP, (long long)NtP * NtP, Nt, Nt, 64, batch,
             0.125f, EPI_NONE, nullptr, nullptr);
        long long rows = (long long)batch * Nt;
        k_softmax<<<(unsigned)((rows + 7) / 8), 256, 0, stream>>>(bigf, probs16, Nt, NtP, rows);
        // attout = P V  (V pre-swizzled, zero-padded K rows)
        gemm(stream, probs16, NtP, (long long)NtP * NtP, v16, (long long)NtP * kHD, 4,
             bigf, nullptr, 64, (long long)NtP * kHD, Nt, 64, NtP, batch,
             1.f, EPI_NONE, nullptr, nullptr);
        k_merge<<<gdiv((long long)Mrows * kDim), 256, 0, stream>>>(bigf, A16a, Nt, NtP, Mrows);
        gemm(stream, A16a, 512, 0, wB_proj + (size_t)i * 512 * 512, 0, 32,
             projOut, nullptr, 512, 0, Mrows, 512, 512, 1, 1.f, EPI_BIAS,
             proj_b + i * 512, nullptr);
    };
    auto mlp_branch = [&](float* src, int Nt, int Mrows, int i, float* scratch) {
        k_ln<<<(Mrows + 7) / 8, 256, 0, stream>>>(src, nullptr, 0, Nt, Mrows,
                                                  n2_w + i * 512, n2_b + i * 512, A16a);
        gemm(stream, A16a, 512, 0, wB_fc1 + (size_t)i * 512 * 512, 0, 32,
             nullptr, A16b, 512, 0, Mrows, 512, 512, 1, 1.f, EPI_BIAS_GELU,
             fc1_b + i * 512, nullptr);
        gemm(stream, A16b, 512, 0, wB_fc2 + (size_t)i * 512 * 512, 0, 32,
             scratch, nullptr, 512, 0, Mrows, 512, 512, 1, 1.f, EPI_BIAS,
             fc2_b + i * 512, nullptr);
        k_add<<<gdiv((long long)Mrows * kDim), 256, 0, stream>>>(src, scratch,
                                                                 (long long)Mrows * kDim);
    };

    for (int i = 0; i < 2; ++i) {
        attn_branch(xs, ys, (long long)kNTY * kDim, kNTX, kNTPX, kMX, i, xproj);
        attn_branch(ys, xs, (long long)kNTX * kDim, kNTY, kNTPY, kMY, i, yproj);
        k_cross<<<gdiv((long long)kMX * kDim), 256, 0, stream>>>(xs, xproj, yproj,
                                                                 kNTX, kNTY, (long long)kMX * kDim);
        k_cross<<<gdiv((long long)kMY * kDim), 256, 0, stream>>>(ys, yproj, xproj,
                                                                 kNTY, kNTX, (long long)kMY * kDim);
        mlp_branch(xs, kNTX, kMX, i, xproj);
        mlp_branch(ys, kNTY, kMY, i, yproj);
    }

    // ---- head ----
    k_headprep<<<(kB * 2 + 7) / 8, 256, 0, stream>>>(xs, ys, norm1_w, norm1_b,
                                                     norm2_w, norm2_b, Ahead);
    gemm(stream, Ahead, 1024, 0, wB_head, 0, 4, out, nullptr, 16, 0,
         kB, 16, 1024, 1, 1.f, EPI_BIAS, head_b, nullptr);
}
